// PairSpatialEncoder_67843303407660
// MI455X (gfx1250) — matrix operand order repaired and checked by reading, not compiled, not run on hardware
//
#include <hip/hip_runtime.h>
#include <stddef.h>

typedef __attribute__((ext_vector_type(2)))  float  v2f;
typedef __attribute__((ext_vector_type(8)))  float  v8f;
typedef __attribute__((ext_vector_type(16))) __bf16 v16bf;

#define KBOX 1024
#define EPSF 1e-6f
#define NROWS (KBOX * (KBOX - 1))        // 1,047,552
#define NTILES (NROWS / 16)              // 65,472 (exact)
#define WAVES_PER_BLOCK 8
#define THREADS (WAVES_PER_BLOCK * 32)
#define BLOCKS 2046                      // 2046*8 waves * 4 tiles = 65,472

// LDS layout.
// W1 (f32, quad-of-K interleave): w1s[((k>>2)*128+n)*4 + (k&3)], k=13 row = b1.
#define W1S_SIZE (4 * 128 * 4)           // 2048 floats
// W2 split bf16 hi/lo, packed K-pairs, B-fragment word order:
//   word[n*68 + K/2] = { bf16(W2[K][n]), bf16(W2[K+1][n]) }   (68 = pad stride)
#define W2B_SIZE (128 * 68)              // 8704 words each (hi, lo)
#define HROW 132                         // padded f32 row stride for h buffer
#define HBUF_SIZE (16 * HROW)            // 2112 floats per wave

__device__ __forceinline__ unsigned pack_hi_bf16(unsigned b0, unsigned b1) {
    return (b1 & 0xFFFF0000u) | (b0 >> 16);     // [15:0]=bf16(x0), [31:16]=bf16(x1)
}

__global__ __launch_bounds__(THREADS)
void pair_mlp_wmma_bf16x(const float* __restrict__ boxes,
                         const float* __restrict__ W1,
                         const float* __restrict__ b1,
                         const float* __restrict__ W2,
                         const float* __restrict__ b2,
                         float* __restrict__ out)
{
    extern __shared__ float smem[];
    unsigned* w2bh   = (unsigned*)smem;              // 8704 words
    unsigned* w2bl   = w2bh + W2B_SIZE;              // 8704 words
    float*    w1s    = (float*)(w2bl + W2B_SIZE);    // 2048 floats
    float*    hbufAll= w1s + W1S_SIZE;               // 8 * 2112 floats

    const int tid = threadIdx.x;

    // ---- stage W2 as split-bf16 packed pairs in B-frag word order ----
    for (int w = tid; w < 64 * 128; w += THREADS) {
        int n = w >> 6;          // column 0..127
        int p = w & 63;          // K/2
        float v0 = W2[(2 * p) * 128 + n];
        float v1 = W2[(2 * p + 1) * 128 + n];
        unsigned u0 = __float_as_uint(v0), u1 = __float_as_uint(v1);
        float l0 = v0 - __uint_as_float(u0 & 0xFFFF0000u);   // exact residual
        float l1 = v1 - __uint_as_float(u1 & 0xFFFF0000u);
        w2bh[n * 68 + p] = pack_hi_bf16(u0, u1);
        w2bl[n * 68 + p] = pack_hi_bf16(__float_as_uint(l0), __float_as_uint(l1));
    }
    // ---- stage W1 (f32), b1 folded into row k=13 ----
    for (int idx = tid; idx < 16 * 128; idx += THREADS) {
        int k = idx >> 7, n = idx & 127;
        float v = 0.0f;
        if (k < 13)       v = W1[idx];
        else if (k == 13) v = b1[n];
        w1s[(((k >> 2) * 128 + n) << 2) + (k & 3)] = v;
    }
    __syncthreads();

    const int lane = tid & 31;
    const int wave = tid >> 5;
    const int m  = lane & 15;   // tile row / n-within-ntile
    const int hh = lane >> 4;   // lane half for WMMA layouts
    float* hb = hbufAll + wave * HBUF_SIZE;

    float bias2[8];
    #pragma unroll
    for (int nt = 0; nt < 8; ++nt) bias2[nt] = b2[nt * 16 + m];

    for (int tile = blockIdx.x * WAVES_PER_BLOCK + wave; tile < NTILES;
         tile += gridDim.x * WAVES_PER_BLOCK) {

        // ---- row -> (i, j) off-diagonal pair ----
        unsigned r = (unsigned)tile * 16u + (unsigned)m;
        unsigned i = r / (KBOX - 1);
        unsigned t = r - i * (KBOX - 1);
        unsigned j = t + (t >= i ? 1u : 0u);

        float4 bi = ((const float4*)boxes)[i];
        float4 bj = ((const float4*)boxes)[j];
        float xi = bi.x, yi = bi.y, wi = bi.z, hi = bi.w;
        float xj = bj.x, yj = bj.y, wj = bj.z, hj = bj.w;

        float inv_wi = 1.0f / wi, inv_hi = 1.0f / hi;
        float f0 = (xj - xi) * inv_wi;
        float f1 = (yj - yi) * inv_hi;
        float f2 = logf(wj * inv_wi + EPSF);
        float f3 = logf(hj * inv_hi + EPSF);
        float x2i = xi + wi, y2i = yi + hi, x2j = xj + wj, y2j = yj + hj;
        float iw = fmaxf(0.0f, fminf(x2i, x2j) - fmaxf(xi, xj));
        float ih = fmaxf(0.0f, fminf(y2i, y2j) - fmaxf(yi, yj));
        float inter = iw * ih;
        float uni   = wi * hi + wj * hj - inter;
        float f8 = inter / (uni + EPSF);

        // layer-1 A fragments (f32 WMMA, k = 4s + 2*hh + {0,1}; k=13 = bias lane)
        v2f aF[4];
        aF[0].x = hh ? f2   : f0;   aF[0].y = hh ? f3   : f1;
        aF[1].x = hh ? wj   : wi;   aF[1].y = hh ? hj   : hi;
        aF[2].x = hh ? yi   : f8;   aF[2].y = hh ? xj   : xi;
        aF[3].x = hh ? 0.0f : yj;   aF[3].y = hh ? 0.0f : 1.0f;

        // ---- layer 1: h = relu(feats @ W1 + b1) -> LDS f32 (row-major) ----
        #pragma unroll
        for (int nt = 0; nt < 8; nt += 2) {
            v8f c0 = {0,0,0,0,0,0,0,0};
            v8f c1 = {0,0,0,0,0,0,0,0};
            #pragma unroll
            for (int s = 0; s < 4; ++s) {
                const v2f bfa = *(const v2f*)&w1s[((s * 128 + nt * 16 + m) << 2) + 2 * hh];
                const v2f bfb = *(const v2f*)&w1s[((s * 128 + (nt + 1) * 16 + m) << 2) + 2 * hh];
                c0 = __builtin_amdgcn_wmma_f32_16x16x4_f32(false, aF[s], false, bfa,
                                                           (short)0, c0, false, false);
                c1 = __builtin_amdgcn_wmma_f32_16x16x4_f32(false, aF[s], false, bfb,
                                                           (short)0, c1, false, false);
            }
            #pragma unroll
            for (int v = 0; v < 8; ++v) {
                hb[(v + 8 * hh) * HROW + nt * 16 + m] =
                    __builtin_amdgcn_fmed3f(c0[v], 0.0f, 3.0e38f);
                hb[(v + 8 * hh) * HROW + (nt + 1) * 16 + m] =
                    __builtin_amdgcn_fmed3f(c1[v], 0.0f, 3.0e38f);
            }
        }

        // ---- build split-bf16 A fragments for layer 2 from h (f32 in LDS) ----
        // 16-bit A 16x32 layout: VGPR v holds K pair (16*hh + 2v) (contiguous k order, same as B).
        v16bf Ahi[4], Alo[4];
        #pragma unroll
        for (int kb = 0; kb < 4; ++kb) {
            union { unsigned u[8]; v16bf v; } Uh, Ul;
            #pragma unroll
            for (int q = 0; q < 4; ++q) {
                int Kb = kb * 32 + 16 * hh + q * 4;
                float4 hv = *(const float4*)&hb[m * HROW + Kb];
                unsigned u0 = __float_as_uint(hv.x), u1 = __float_as_uint(hv.y);
                unsigned u2 = __float_as_uint(hv.z), u3 = __float_as_uint(hv.w);
                Uh.u[2 * q + 0] = pack_hi_bf16(u0, u1);
                Uh.u[2 * q + 1] = pack_hi_bf16(u2, u3);
                float l0 = hv.x - __uint_as_float(u0 & 0xFFFF0000u);
                float l1 = hv.y - __uint_as_float(u1 & 0xFFFF0000u);
                float l2 = hv.z - __uint_as_float(u2 & 0xFFFF0000u);
                float l3 = hv.w - __uint_as_float(u3 & 0xFFFF0000u);
                Ul.u[2 * q + 0] = pack_hi_bf16(__float_as_uint(l0), __float_as_uint(l1));
                Ul.u[2 * q + 1] = pack_hi_bf16(__float_as_uint(l2), __float_as_uint(l3));
            }
            Ahi[kb] = Uh.v; Alo[kb] = Ul.v;
        }

        // ---- layer 2: out = h @ W2 + b2 via split-bf16 WMMA (2 chains) ----
        float* outp = out + (size_t)tile * (16 * 128);
        #pragma unroll
        for (int nt = 0; nt < 8; nt += 2) {
            v8f c0, c1;
            #pragma unroll
            for (int v = 0; v < 8; ++v) { c0[v] = bias2[nt]; c1[v] = bias2[nt + 1]; }
            #pragma unroll
            for (int kb = 0; kb < 4; ++kb) {
                union { uint4 q[2]; v16bf v; } B0h, B0l, B1h, B1l;
                int na = (nt * 16 + m) * 68 + 16 * kb + 8 * hh;
                int nb = ((nt + 1) * 16 + m) * 68 + 16 * kb + 8 * hh;
                B0h.q[0] = *(const uint4*)&w2bh[na];     B0h.q[1] = *(const uint4*)&w2bh[na + 4];
                B1h.q[0] = *(const uint4*)&w2bh[nb];     B1h.q[1] = *(const uint4*)&w2bh[nb + 4];
                B0l.q[0] = *(const uint4*)&w2bl[na];     B0l.q[1] = *(const uint4*)&w2bl[na + 4];
                B1l.q[0] = *(const uint4*)&w2bl[nb];     B1l.q[1] = *(const uint4*)&w2bl[nb + 4];
                c0 = __builtin_amdgcn_wmma_f32_16x16x32_bf16(false, Ahi[kb], false, B0h.v,
                                                             (short)0, c0, false, false);
                c1 = __builtin_amdgcn_wmma_f32_16x16x32_bf16(false, Ahi[kb], false, B1h.v,
                                                             (short)0, c1, false, false);
                c0 = __builtin_amdgcn_wmma_f32_16x16x32_bf16(false, Alo[kb], false, B0h.v,
                                                             (short)0, c0, false, false);
                c1 = __builtin_amdgcn_wmma_f32_16x16x32_bf16(false, Alo[kb], false, B1h.v,
                                                             (short)0, c1, false, false);
                c0 = __builtin_amdgcn_wmma_f32_16x16x32_bf16(false, Ahi[kb], false, B0l.v,
                                                             (short)0, c0, false, false);
                c1 = __builtin_amdgcn_wmma_f32_16x16x32_bf16(false, Ahi[kb], false, B1l.v,
                                                             (short)0, c1, false, false);
            }
            #pragma unroll
            for (int v = 0; v < 8; ++v) {
                __builtin_nontemporal_store(c0[v], &outp[(v + 8 * hh) * 128 + nt * 16 + m]);
                __builtin_nontemporal_store(c1[v], &outp[(v + 8 * hh) * 128 + (nt + 1) * 16 + m]);
            }
        }
    }
}

extern "C" void kernel_launch(void* const* d_in, const int* in_sizes, int n_in,
                              void* d_out, int out_size, void* d_ws, size_t ws_size,
                              hipStream_t stream) {
    const float* boxes = (const float*)d_in[0];
    const float* W1    = (const float*)d_in[1];
    const float* b1    = (const float*)d_in[2];
    const float* W2    = (const float*)d_in[3];
    const float* b2    = (const float*)d_in[4];
    float* out = (float*)d_out;

    size_t shmem = (size_t)(2 * W2B_SIZE + W1S_SIZE +
                            WAVES_PER_BLOCK * HBUF_SIZE) * sizeof(float);
    pair_mlp_wmma_bf16x<<<BLOCKS, THREADS, shmem, stream>>>(boxes, W1, b1, W2, b2, out);
}
